// MultiScaleTimeMixer_1949915152742
// MI455X (gfx1250) — matrix operand order repaired and checked by reading, not compiled, not run on hardware
//
#include <hip/hip_runtime.h>

// ---------------------------------------------------------------------------
// MultiScaleTimeMixer for MI455X (gfx1250), wave32 + bf16 WMMA.
// conv1/conv2 as GEMMs, per-batch LN, TriU masked GEMMs with hardswish.
// TriU GEMMs: B strip staged per-block in LDS via async global->LDS loads.
// ---------------------------------------------------------------------------

typedef __attribute__((ext_vector_type(16))) __bf16 v16bf;
typedef __attribute__((ext_vector_type(8)))  float  v8f;

#define B_   128
#define T_   256
#define C_   256
#define T1_  128
#define T2_  64
#define TT_  448   // T + T1 + T2
#define LDSROW_ 456 // TT_ + 8 ushorts padding -> conflict-free ds_load_b128

union FragBF {
    v16bf v;
    uint4 q[2];  // 2 x 16 bytes = 32 bytes = 16 bf16
};

__device__ __forceinline__ unsigned short f2bf(float f) {
    union { float f; unsigned int u; } cv; cv.f = f;
    unsigned int u = cv.u;
    unsigned int r = 0x7FFFu + ((u >> 16) & 1u);   // round-to-nearest-even
    return (unsigned short)((u + r) >> 16);
}

// ---------------------------------------------------------------------------
// Prep kernels (memory-bound, elementwise)
// ---------------------------------------------------------------------------

// x: [B,T,C] f32 -> Xbf [B,T,C] bf16 (conv GEMM A) and A [b*C+c][TT] bf16 (TriU A, t<256)
__global__ void k_x_prep(const float* __restrict__ X,
                         unsigned short* __restrict__ Xbf,
                         unsigned short* __restrict__ A) {
    int idx = blockIdx.x * blockDim.x + threadIdx.x;
    if (idx >= B_ * T_ * C_) return;
    unsigned short v = f2bf(X[idx]);
    Xbf[idx] = v;
    int c = idx & (C_ - 1);
    int t = (idx >> 8) & (T_ - 1);
    int b = idx >> 16;
    A[((size_t)(b * C_ + c)) * TT_ + t] = v;
}

// conv weights [co][ci][K] f32 -> Wt [co][k*C+ci] bf16 (B-matrix: N-major rows, K contiguous)
__global__ void k_convw_prep(const float* __restrict__ W,
                             unsigned short* __restrict__ Wt, int Kk) {
    int idx = blockIdx.x * blockDim.x + threadIdx.x;
    if (idx >= C_ * C_ * Kk) return;
    int k  = idx % Kk;
    int ci = (idx / Kk) & (C_ - 1);
    int co = idx / (Kk * C_);
    Wt[(size_t)co * (C_ * Kk) + k * C_ + ci] = f2bf(W[idx]);
}

// triu weights [k][n] f32 -> WmT [n][k] bf16 with mask (keep k <= n)
__global__ void k_triu_prep(const float* __restrict__ W,
                            unsigned short* __restrict__ Wt) {
    int idx = blockIdx.x * blockDim.x + threadIdx.x;
    if (idx >= TT_ * TT_) return;
    int k = idx / TT_;
    int n = idx % TT_;
    Wt[(size_t)n * TT_ + k] = (k <= n) ? f2bf(W[idx]) : (unsigned short)0;
}

// ---------------------------------------------------------------------------
// Conv as WMMA GEMM: Y[(b,t'),co] = bias[co] + sum_kk Xbf[row(b,t'), kk] * Wt[co, kk]
// One wave computes a 16(M) x 64(N) tile. Kc = K*C (512 or 1024).
// ---------------------------------------------------------------------------
__global__ __launch_bounds__(256)
void k_conv_gemm(const unsigned short* __restrict__ Xbf,
                 const unsigned short* __restrict__ Wt,
                 const float* __restrict__ bias,
                 float* __restrict__ Y,
                 int tshift, int sshift, int Kc) {
    int wave = (blockIdx.x * blockDim.x + threadIdx.x) >> 5;
    int lane = threadIdx.x & 31;
    int mTile = wave >> 2;           // N=256 -> 4 quads of 64
    int nq    = wave & 3;
    int m0 = mTile * 16, n0 = nq * 64;
    int lm = lane & 15, lh = lane >> 4;

    int mm = m0 + lm;
    int bb = mm >> tshift;
    int tt = mm & ((1 << tshift) - 1);
    const unsigned short* arow  = Xbf + (((size_t)(bb << 8) + ((size_t)tt << sshift)) << 8) + lh * 8;
    const unsigned short* brow0 = Wt + (size_t)(n0 + lm) * Kc + lh * 8;

    v8f z = {0.f,0.f,0.f,0.f,0.f,0.f,0.f,0.f};
    v8f acc[4] = {z, z, z, z};

    for (int k0 = 0; k0 < Kc; k0 += 32) {
        FragBF a;
        a.q[0] = *(const uint4*)(arow + k0);
        a.q[1] = *(const uint4*)(arow + k0 + 16);
#pragma unroll
        for (int j = 0; j < 4; ++j) {
            FragBF bfrag;
            const unsigned short* br = brow0 + (size_t)j * 16 * Kc + k0;
            bfrag.q[0] = *(const uint4*)(br);
            bfrag.q[1] = *(const uint4*)(br + 16);
            acc[j] = __builtin_amdgcn_wmma_f32_16x16x32_bf16(
                false, a.v, false, bfrag.v, (short)0, acc[j], false, false);
        }
    }
#pragma unroll
    for (int j = 0; j < 4; ++j) {
        int n = n0 + j * 16 + lm;
        float bs = bias[n];
#pragma unroll
        for (int r = 0; r < 8; ++r) {
            int m = m0 + lh * 8 + r;
            Y[(size_t)m * C_ + n] = acc[j][r] + bs;
        }
    }
}

// ---------------------------------------------------------------------------
// LayerNorm over whole (T',C) plane per batch: stats then apply+transpose
// ---------------------------------------------------------------------------
__global__ __launch_bounds__(256)
void k_ln_stats(const float* __restrict__ Y, float* __restrict__ stats, int Nelem) {
    int b = blockIdx.x;
    const float* p = Y + (size_t)b * Nelem;
    float s = 0.f, ss = 0.f;
    for (int i = threadIdx.x; i < Nelem; i += 256) {
        float v = p[i]; s += v; ss += v * v;
    }
    __shared__ float sh[256], sh2[256];
    sh[threadIdx.x] = s; sh2[threadIdx.x] = ss;
    __syncthreads();
    for (int st = 128; st > 0; st >>= 1) {
        if (threadIdx.x < st) {
            sh[threadIdx.x]  += sh[threadIdx.x + st];
            sh2[threadIdx.x] += sh2[threadIdx.x + st];
        }
        __syncthreads();
    }
    if (threadIdx.x == 0) {
        float mean = sh[0] / (float)Nelem;
        float var  = sh2[0] / (float)Nelem - mean * mean;
        stats[b * 2]     = mean;
        stats[b * 2 + 1] = rsqrtf(var + 1e-5f);
    }
}

// apply LN and scatter (transposed) into TriU A-matrix at time offset toff
__global__ void k_ln_apply_tr(const float* __restrict__ Y, const float* __restrict__ stats,
                              const float* __restrict__ g, const float* __restrict__ bet,
                              unsigned short* __restrict__ A, int Tsub, int toff) {
    int idx = blockIdx.x * blockDim.x + threadIdx.x;
    int total = B_ * Tsub * C_;
    if (idx >= total) return;
    int c = idx & (C_ - 1);
    int t = (idx >> 8) % Tsub;
    int b = idx / (Tsub * C_);
    float mu = stats[b * 2], rs = stats[b * 2 + 1];
    float v = (Y[idx] - mu) * rs * g[t * C_ + c] + bet[t * C_ + c];
    A[((size_t)(b * C_ + c)) * TT_ + toff + t] = f2bf(v);
}

// ---------------------------------------------------------------------------
// TriU GEMM (EPI=0: +bias, hardswish, bf16 out [m][448];
//            EPI=1: +bias, fp32 scatter to [B,TT,C])
// Block = 256 thr (8 waves); wave computes 32(M) x 64(N); block = 256 x 64.
// B strip [64 x 448] staged in LDS once per block via async global->LDS.
// ---------------------------------------------------------------------------
template <int EPI>
__global__ __launch_bounds__(256)
void k_gemm_triu(const unsigned short* __restrict__ A,
                 const unsigned short* __restrict__ Bt,
                 const float* __restrict__ bias,
                 void* __restrict__ OutV) {
    __shared__ unsigned short shB[64 * LDSROW_];   // 58,368 bytes

    int wid  = threadIdx.x >> 5;
    int lane = threadIdx.x & 31;
    int lm = lane & 15, lh = lane >> 4;

    int mblk = blockIdx.x / 7;
    int nq   = blockIdx.x % 7;
    int n0 = nq * 64;
    int m0 = mblk * 256 + wid * 32;

    // --- stage B strip (rows n0..n0+63 of Bt) into LDS with async copies ---
    {
        const unsigned short* src = Bt + (size_t)n0 * TT_;
        // 64 rows * 56 b128-chunks = 3584 transfers; 256 threads -> 14 each
        for (int i = threadIdx.x; i < 64 * 56; i += 256) {
            int row = i / 56;
            int col = i - row * 56;
            unsigned int ldsOff = (unsigned int)(row * (LDSROW_ * 2) + col * 16);
            const unsigned short* g = src + row * TT_ + col * 8;
            asm volatile("global_load_async_to_lds_b128 %0, %1, off"
                         :: "v"(ldsOff), "v"(g) : "memory");
        }
#if __has_builtin(__builtin_amdgcn_s_wait_asynccnt)
        __builtin_amdgcn_s_wait_asynccnt(0);
#else
        asm volatile("s_wait_asynccnt 0x0" ::: "memory");
#endif
        __syncthreads();
    }

    const unsigned short* arow0 = A + (size_t)(m0 + lm) * TT_ + lh * 8;
    const unsigned short* arow1 = arow0 + (size_t)16 * TT_;

    v8f z = {0.f,0.f,0.f,0.f,0.f,0.f,0.f,0.f};
    v8f acc[8] = {z, z, z, z, z, z, z, z};

    for (int k0 = 0; k0 < TT_; k0 += 32) {
        FragBF a0, a1;
        a0.q[0] = *(const uint4*)(arow0 + k0);
        a0.q[1] = *(const uint4*)(arow0 + k0 + 16);
        a1.q[0] = *(const uint4*)(arow1 + k0);
        a1.q[1] = *(const uint4*)(arow1 + k0 + 16);
#pragma unroll
        for (int j = 0; j < 4; ++j) {
            FragBF bfrag;
            const unsigned short* ls = &shB[(j * 16 + lm) * LDSROW_ + k0 + lh * 8];
            bfrag.q[0] = *(const uint4*)(ls);
            bfrag.q[1] = *(const uint4*)(ls + 16);
            acc[j] = __builtin_amdgcn_wmma_f32_16x16x32_bf16(
                false, a0.v, false, bfrag.v, (short)0, acc[j], false, false);
            acc[4 + j] = __builtin_amdgcn_wmma_f32_16x16x32_bf16(
                false, a1.v, false, bfrag.v, (short)0, acc[4 + j], false, false);
        }
    }

#pragma unroll
    for (int mi = 0; mi < 2; ++mi) {
        int mbase = m0 + mi * 16 + lh * 8;
#pragma unroll
        for (int j = 0; j < 4; ++j) {
            int n = n0 + j * 16 + lm;
            float bs = bias[n];
            v8f a = acc[mi * 4 + j];
#pragma unroll
            for (int r = 0; r < 8; ++r) {
                int m = mbase + r;
                float v = a[r] + bs;
                if (EPI == 0) {
                    float hs = v * fminf(fmaxf(v + 3.f, 0.f), 6.f) * (1.f / 6.f);
                    ((unsigned short*)OutV)[(size_t)m * TT_ + n] = f2bf(hs);
                } else {
                    int b = m >> 8;            // m = b*C + c
                    int c = m & (C_ - 1);
                    ((float*)OutV)[((size_t)b * TT_ + n) * C_ + c] = v;
                }
            }
        }
    }
}

// ---------------------------------------------------------------------------
// Launch
// ---------------------------------------------------------------------------
extern "C" void kernel_launch(void* const* d_in, const int* in_sizes, int n_in,
                              void* d_out, int out_size, void* d_ws, size_t ws_size,
                              hipStream_t stream) {
    const float* x       = (const float*)d_in[0];
    const float* conv_w1 = (const float*)d_in[1];
    const float* conv_b1 = (const float*)d_in[2];
    const float* conv_w2 = (const float*)d_in[3];
    const float* conv_b2 = (const float*)d_in[4];
    const float* ln1_g   = (const float*)d_in[5];
    const float* ln1_b   = (const float*)d_in[6];
    const float* ln2_g   = (const float*)d_in[7];
    const float* ln2_b   = (const float*)d_in[8];
    const float* triu_w1 = (const float*)d_in[9];
    const float* triu_b1 = (const float*)d_in[10];
    const float* triu_w2 = (const float*)d_in[11];
    const float* triu_b2 = (const float*)d_in[12];
    float* out = (float*)d_out;
    char* ws = (char*)d_ws;

    // workspace layout (bytes)
    const size_t XBF_OFF = 0;                                    // 16,777,216
    const size_t A_OFF   = XBF_OFF + (size_t)B_*T_*C_*2;         // 29,360,128
    const size_t A2_OFF  = A_OFF   + (size_t)B_*C_*TT_*2;
    const size_t Y1_OFF  = A2_OFF  + (size_t)B_*C_*TT_*2;        // 16,777,216
    const size_t Y2_OFF  = Y1_OFF  + (size_t)B_*T1_*C_*4;        //  8,388,608
    const size_t WC1_OFF = Y2_OFF  + (size_t)B_*T2_*C_*4;        //    262,144
    const size_t WC2_OFF = WC1_OFF + (size_t)C_*C_*2*2;          //    524,288
    const size_t WT1_OFF = WC2_OFF + (size_t)C_*C_*4*2;          //    401,408
    const size_t WT2_OFF = WT1_OFF + (size_t)TT_*TT_*2;
    const size_t ST1_OFF = WT2_OFF + (size_t)TT_*TT_*2;
    const size_t ST2_OFF = ST1_OFF + 1024;

    unsigned short* Xbf  = (unsigned short*)(ws + XBF_OFF);
    unsigned short* A    = (unsigned short*)(ws + A_OFF);
    unsigned short* A2   = (unsigned short*)(ws + A2_OFF);
    float*          Y1   = (float*)(ws + Y1_OFF);
    float*          Y2   = (float*)(ws + Y2_OFF);
    unsigned short* Wc1T = (unsigned short*)(ws + WC1_OFF);
    unsigned short* Wc2T = (unsigned short*)(ws + WC2_OFF);
    unsigned short* Wm1T = (unsigned short*)(ws + WT1_OFF);
    unsigned short* Wm2T = (unsigned short*)(ws + WT2_OFF);
    float*          St1  = (float*)(ws + ST1_OFF);
    float*          St2  = (float*)(ws + ST2_OFF);

    const int TPB = 256;
    auto cdiv = [](int a, int b) { return (a + b - 1) / b; };

    // --- prep ---
    k_x_prep<<<cdiv(B_*T_*C_, TPB), TPB, 0, stream>>>(x, Xbf, A);
    k_convw_prep<<<cdiv(C_*C_*2, TPB), TPB, 0, stream>>>(conv_w1, Wc1T, 2);
    k_convw_prep<<<cdiv(C_*C_*4, TPB), TPB, 0, stream>>>(conv_w2, Wc2T, 4);
    k_triu_prep<<<cdiv(TT_*TT_, TPB), TPB, 0, stream>>>(triu_w1, Wm1T);
    k_triu_prep<<<cdiv(TT_*TT_, TPB), TPB, 0, stream>>>(triu_w2, Wm2T);

    // --- conv GEMMs: waves = (M/16)*(N/64), 8 waves/block ---
    {
        int waves1 = (B_*T1_/16) * (C_/64);   // 4096
        k_conv_gemm<<<waves1/8, TPB, 0, stream>>>(Xbf, Wc1T, conv_b1, Y1, 7, 1, 2*C_);
        int waves2 = (B_*T2_/16) * (C_/64);   // 2048
        k_conv_gemm<<<waves2/8, TPB, 0, stream>>>(Xbf, Wc2T, conv_b2, Y2, 6, 2, 4*C_);
    }

    // --- layernorm ---
    k_ln_stats<<<B_, TPB, 0, stream>>>(Y1, St1, T1_*C_);
    k_ln_stats<<<B_, TPB, 0, stream>>>(Y2, St2, T2_*C_);
    k_ln_apply_tr<<<cdiv(B_*T1_*C_, TPB), TPB, 0, stream>>>(Y1, St1, ln1_g, ln1_b, A, T1_, T_);
    k_ln_apply_tr<<<cdiv(B_*T2_*C_, TPB), TPB, 0, stream>>>(Y2, St2, ln2_g, ln2_b, A, T2_, T_+T1_);

    // --- TriU GEMMs: blocks = (32768/256) * 7 = 896, each 8 waves x (32x64) ---
    {
        int blocks = (B_*C_/256) * (TT_/64);
        k_gemm_triu<0><<<blocks, TPB, 0, stream>>>(A,  Wm1T, triu_b1, (void*)A2);
        k_gemm_triu<1><<<blocks, TPB, 0, stream>>>(A2, Wm2T, triu_b2, (void*)out);
    }
}
